// TransformerEncoder_1632087573147
// MI455X (gfx1250) — compile-verified, matrix operand order checked by reading, not run on hardware
//
#include <hip/hip_runtime.h>
#include <hip/hip_bf16.h>

typedef unsigned int u32;
typedef unsigned short u16;

typedef __attribute__((ext_vector_type(16))) __bf16 v16bf;
typedef __attribute__((ext_vector_type(8)))  float  v8f;

namespace {
constexpr int    Sn  = 3, Bn = 32, Ln = 256, En = 512, Fn = 2048, Hn = 16, DHn = 32;
constexpr size_t BLn = (size_t)Bn * Ln;   // 8192 rows
constexpr int    HBn = Hn * Bn;           // 512 softmax-reduction depth
constexpr size_t LLn = (size_t)Ln * Ln;   // 65536
}

// ---------------- bf16 helpers ----------------
// Hardware RNE convert (v_fma_mixlo/hi_bf16 or v_cvt_pk_bf16_f32 when vectorized).
__device__ __forceinline__ u16 f2bf(float f) {
  return __builtin_bit_cast(u16, (__bf16)f);
}
__device__ __forceinline__ float bf2f(u16 h) {
  u32 u = ((u32)h) << 16;
  return __builtin_bit_cast(float, u);
}

// ---------------- CDNA5 async global->LDS copy (ASYNCcnt) ----------------
// VDST = LDS byte offset in a VGPR (low 32 bits of generic LDS pointer),
// VADDR = 64-bit global address. Tracked by ASYNCcnt, waited via s_wait_asynccnt.
__device__ __forceinline__ void async_ld_b128(u32 lds_off, const void* gaddr) {
  asm volatile("global_load_async_to_lds_b128 %0, %1, off"
               :: "v"(lds_off), "v"(gaddr)
               : "memory");
}
__device__ __forceinline__ void wait_async0() {
  asm volatile("s_wait_asynccnt 0" ::: "memory");
}

// ---------------- WMMA fragment loaders ----------------
// A/B bf16 16x32 fragment (ISA 7.12.2): lane = (M or N) mod 16, K-half = lane>>4.
// LDS tiles row-major, 32 bf16 (=4 uint4) per row, padded to 5 uint4/row for banks.
struct FragQ { uint4 lo, hi; };
struct FragW { u32 u[8]; };

__device__ __forceinline__ v16bf ldfrag(const uint4* t, int row, int half) {
  FragQ f;
  f.lo = t[row * 5 + half];
  f.hi = t[row * 5 + 2 + half];
  return __builtin_bit_cast(v16bf, f);
}

// B fragment gathered from an LDS tile stored [k][n] as u16 (ldn columns).
__device__ __forceinline__ v16bf ldfrag_kxn(const u16* t, int n, int half, int ldn) {
  FragW f;
#pragma unroll
  for (int j = 0; j < 8; ++j) {
    int kd = (j < 4) ? (half * 4 + j) : (8 + half * 4 + (j - 4));
    int k  = kd * 2;
    f.u[j] = (u32)t[k * ldn + n] | ((u32)t[(k + 1) * ldn + n] << 16);
  }
  return __builtin_bit_cast(v16bf, f);
}

__device__ __forceinline__ v8f wmma_bf16(v16bf a, v16bf b, v8f c) {
  return __builtin_amdgcn_wmma_f32_16x16x32_bf16(false, a, false, b, (short)0, c,
                                                 false, false);
}

// ---------------- generic bf16 GEMM:  C[M,N] = act(A[MxK] * W[NxK]^T + bias) ----------------
// 256 threads = 8 waves (4x2), BM=128 BN=128 BK=32; each wave 32x64 via 2x4 WMMA.
// Ping-pong LDS buffers; next K-tile streamed global->LDS by the async DMA path
// (ASYNCcnt) while the 8 WMMAs of the current tile execute.
template <bool RELU>
__global__ __launch_bounds__(256) void gemm_bf16(
    const u16* __restrict__ A, const u16* __restrict__ W,
    const float* __restrict__ bias, float* __restrict__ outF,
    u16* __restrict__ outB, int M, int N, int K) {
  __shared__ uint4 As[2][128 * 5];
  __shared__ uint4 Bs[2][128 * 5];
  const int tid = threadIdx.x;
  const int m0 = blockIdx.y * 128, n0 = blockIdx.x * 128;
  const int lane = tid & 31, w = tid >> 5;
  const int wm = (w >> 1) * 32, wn = (w & 1) * 64;
  const int half = lane >> 4, l16 = lane & 15;
  const int lr = tid >> 2, lq = tid & 3;   // cooperative copy: row 0..63(+64), quad

  v8f acc[2][4] = {{{}, {}, {}, {}}, {{}, {}, {}, {}}};

  auto issue_tile = [&](int k0, int buf) {
    async_ld_b128((u32)(size_t)&As[buf][lr * 5 + lq],
                  (const uint4*)(A + (size_t)(m0 + lr) * K + k0) + lq);
    async_ld_b128((u32)(size_t)&As[buf][(lr + 64) * 5 + lq],
                  (const uint4*)(A + (size_t)(m0 + 64 + lr) * K + k0) + lq);
    async_ld_b128((u32)(size_t)&Bs[buf][lr * 5 + lq],
                  (const uint4*)(W + (size_t)(n0 + lr) * K + k0) + lq);
    async_ld_b128((u32)(size_t)&Bs[buf][(lr + 64) * 5 + lq],
                  (const uint4*)(W + (size_t)(n0 + 64 + lr) * K + k0) + lq);
  };

  issue_tile(0, 0);
  wait_async0();
  __syncthreads();

  int cur = 0;
  for (int k0 = 0; k0 < K; k0 += 32) {
    const bool more = (k0 + 32) < K;
    if (more) {
      issue_tile(k0 + 32, cur ^ 1);
      if (k0 + 64 < K) {  // pull the tile after next toward L2 early
        __builtin_prefetch(A + (size_t)(m0 + lr) * K + k0 + 64, 0, 1);
        __builtin_prefetch(W + (size_t)(n0 + lr) * K + k0 + 64, 0, 1);
      }
    }
    v16bf af0 = ldfrag(As[cur], wm + l16, half);
    v16bf af1 = ldfrag(As[cur], wm + 16 + l16, half);
    v16bf bf0 = ldfrag(Bs[cur], wn + l16, half);
    v16bf bf1 = ldfrag(Bs[cur], wn + 16 + l16, half);
    v16bf bf2 = ldfrag(Bs[cur], wn + 32 + l16, half);
    v16bf bf3 = ldfrag(Bs[cur], wn + 48 + l16, half);
    acc[0][0] = wmma_bf16(af0, bf0, acc[0][0]);
    acc[0][1] = wmma_bf16(af0, bf1, acc[0][1]);
    acc[0][2] = wmma_bf16(af0, bf2, acc[0][2]);
    acc[0][3] = wmma_bf16(af0, bf3, acc[0][3]);
    acc[1][0] = wmma_bf16(af1, bf0, acc[1][0]);
    acc[1][1] = wmma_bf16(af1, bf1, acc[1][1]);
    acc[1][2] = wmma_bf16(af1, bf2, acc[1][2]);
    acc[1][3] = wmma_bf16(af1, bf3, acc[1][3]);
    if (more) wait_async0();   // this wave's DMA into buf^1 done
    __syncthreads();           // all waves: done reading cur + done DMA
    cur ^= 1;
  }

#pragma unroll
  for (int i = 0; i < 2; ++i)
#pragma unroll
    for (int j = 0; j < 4; ++j) {
      int n = n0 + wn + j * 16 + l16;
      float bv = bias ? bias[n] : 0.f;
#pragma unroll
      for (int r = 0; r < 8; ++r) {
        int m = m0 + wm + i * 16 + r + 8 * half;  // C layout: VGPR r -> M=r+8*(lane>>4)
        float v = acc[i][j][r] + bv;
        if (RELU) v = fmaxf(v, 0.f);
        size_t o = (size_t)m * N + n;
        if (outF) outF[o] = v;
        if (outB) outB[o] = f2bf(v);
      }
    }
}

// ---------------- attention scores: S[hb,q,k] = (Q_h . K_h)/sqrt(dh), bf16 out ----------------
// grid (4,4,HB), 128 threads = 4 waves (2x2), 64x64 tile, K=32 = single WMMA step.
__global__ __launch_bounds__(128) void attn_scores(const u16* __restrict__ Q,
                                                   const u16* __restrict__ Kmat,
                                                   u16* __restrict__ Sc) {
  __shared__ uint4 Qs[64 * 5];
  __shared__ uint4 Ks[64 * 5];
  const int hb = blockIdx.z, h = hb >> 5, b = hb & 31;  // hb = h*B + b, B=32
  const int m0 = blockIdx.y * 64, n0 = blockIdx.x * 64;
  const int tid = threadIdx.x;
#pragma unroll
  for (int it = 0; it < 2; ++it) {
    int idx = tid + it * 128;
    int row = idx >> 2, q = idx & 3;
    Qs[row * 5 + q] =
        *((const uint4*)(Q + (size_t)(b * Ln + m0 + row) * En + h * DHn) + q);
    Ks[row * 5 + q] =
        *((const uint4*)(Kmat + (size_t)(b * Ln + n0 + row) * En + h * DHn) + q);
  }
  __syncthreads();
  const int lane = tid & 31, w = tid >> 5;
  const int wm = (w >> 1) * 32, wn = (w & 1) * 32;
  const int half = lane >> 4, l16 = lane & 15;
  v16bf a0 = ldfrag(Qs, wm + l16, half);
  v16bf a1 = ldfrag(Qs, wm + 16 + l16, half);
  v16bf b0 = ldfrag(Ks, wn + l16, half);
  v16bf b1 = ldfrag(Ks, wn + 16 + l16, half);
  v8f acc[2][2] = {{{}, {}}, {{}, {}}};
  acc[0][0] = wmma_bf16(a0, b0, acc[0][0]);
  acc[0][1] = wmma_bf16(a0, b1, acc[0][1]);
  acc[1][0] = wmma_bf16(a1, b0, acc[1][0]);
  acc[1][1] = wmma_bf16(a1, b1, acc[1][1]);
  const float scale = 0.17677669529663687f;  // 1/sqrt(32)
#pragma unroll
  for (int i = 0; i < 2; ++i)
#pragma unroll
    for (int j = 0; j < 2; ++j) {
      int n = n0 + wn + j * 16 + l16;
#pragma unroll
      for (int r = 0; r < 8; ++r) {
        int m = m0 + wm + i * 16 + r + 8 * half;
        Sc[(size_t)hb * LLn + (size_t)m * Ln + n] = f2bf(acc[i][j][r] * scale);
      }
    }
}

// ---------------- legacy softmax over dim0 of (H*B, L, L) ----------------
// one thread per (q,k); serial online-softmax over the 512 hb values; bf16 out.
__global__ __launch_bounds__(256) void softmax_hb(const u16* __restrict__ S,
                                                  u16* __restrict__ Aout) {
  const size_t base = (size_t)blockIdx.x * 256 + threadIdx.x;  // 0..65535
  float m = -3.0e38f, s = 0.f;
  for (int hb = 0; hb < HBn; ++hb) {
    float x = bf2f(S[(size_t)hb * LLn + base]);
    float nm = fmaxf(m, x);
    s = s * __expf(m - nm) + __expf(x - nm);
    m = nm;
  }
  const float inv = 1.f / s;
  for (int hb = 0; hb < HBn; ++hb) {
    float x = bf2f(S[(size_t)hb * LLn + base]);
    Aout[(size_t)hb * LLn + base] = f2bf(__expf(x - m) * inv);
  }
}

// ---------------- attn @ V: per hb, C(256x32) = Attn(256x256) * V_h(256x32) ----------------
// grid (4, HB), 128 threads = 4 waves, each wave one 16-row strip (1x2 WMMA).
__global__ __launch_bounds__(128) void attn_av(const u16* __restrict__ Attn,
                                               const u16* __restrict__ V,
                                               float* __restrict__ O) {
  __shared__ uint4 As[64 * 5];   // attn tile 64x32
  __shared__ u32   Vs[32 * 16];  // V tile [k][d] 32x32 bf16
  const int hb = blockIdx.y, h = hb >> 5, b = hb & 31;
  const int m0 = blockIdx.x * 64;
  const int tid = threadIdx.x;
  const int lane = tid & 31, w = tid >> 5;
  const int half = lane >> 4, l16 = lane & 15;
  v8f acc[2] = {{}, {}};

  for (int k0 = 0; k0 < Ln; k0 += 32) {
#pragma unroll
    for (int it = 0; it < 2; ++it) {
      int idx = tid + it * 128;
      int row = idx >> 2, q = idx & 3;
      As[row * 5 + q] =
          *((const uint4*)(Attn + (size_t)hb * LLn + (size_t)(m0 + row) * Ln + k0) + q);
    }
#pragma unroll
    for (int it = 0; it < 4; ++it) {  // 32 k-rows x 16 dwords
      int j = tid + it * 128;
      int r = j >> 4, c = j & 15;
      Vs[j] = *((const u32*)(V + (size_t)(b * Ln + k0 + r) * En + h * DHn) + c);
    }
    __syncthreads();
    v16bf a   = ldfrag(As, w * 16 + l16, half);
    v16bf vb0 = ldfrag_kxn((const u16*)Vs, l16, half, 32);
    v16bf vb1 = ldfrag_kxn((const u16*)Vs, 16 + l16, half, 32);
    acc[0] = wmma_bf16(a, vb0, acc[0]);
    acc[1] = wmma_bf16(a, vb1, acc[1]);
    __syncthreads();
  }
#pragma unroll
  for (int j = 0; j < 2; ++j) {
    int n = j * 16 + l16;
#pragma unroll
    for (int r = 0; r < 8; ++r) {
      int m = m0 + w * 16 + r + 8 * half;
      O[(size_t)(b * Ln + m) * En + h * DHn + n] = acc[j][r];
    }
  }
}

// ---------------- BatchNorm (per column over all BLn rows; optional fused residual) ----------------
__global__ __launch_bounds__(256) void bn_stats(const float* __restrict__ X,
                                                const float* __restrict__ R,
                                                float* __restrict__ st) {
  __shared__ float ssum[256], ssq[256];
  const int tc = threadIdx.x & 31, tr = threadIdx.x >> 5;
  const int c = blockIdx.x * 32 + tc;
  float s = 0.f, q = 0.f;
  for (int r = tr; r < (int)BLn; r += 8) {
    size_t i = (size_t)r * En + c;
    float xv = X[i];
    if (R) xv += R[i];
    s += xv;
    q += xv * xv;
  }
  ssum[threadIdx.x] = s;
  ssq[threadIdx.x] = q;
  __syncthreads();
  if (tr == 0) {
    for (int t = 1; t < 8; ++t) { s += ssum[t * 32 + tc]; q += ssq[t * 32 + tc]; }
    float mean = s * (1.f / (float)BLn);
    float var  = q * (1.f / (float)BLn) - mean * mean;   // biased, as torch BN
    st[c] = mean;
    st[En + c] = rsqrtf(var + 1e-5f);
  }
}

__global__ __launch_bounds__(256) void bn_apply(
    const float* __restrict__ X, const float* __restrict__ R,
    const float* __restrict__ g, const float* __restrict__ bta,
    const float* __restrict__ st, float* __restrict__ outF,
    u16* __restrict__ outB) {
  const size_t total = BLn * En;
  const size_t stride = (size_t)gridDim.x * blockDim.x;
  for (size_t i = (size_t)blockIdx.x * blockDim.x + threadIdx.x; i < total; i += stride) {
    int c = (int)(i & (size_t)(En - 1));
    float xv = X[i];
    if (R) xv += R[i];
    float y = g[c] * (xv - st[c]) * st[En + c] + bta[c];
    outF[i] = y;
    if (outB) outB[i] = f2bf(y);
  }
}

// ---------------- casts / copy ----------------
__global__ void cast_f2b(const float* __restrict__ in, u16* __restrict__ out, size_t n) {
  const size_t stride = (size_t)gridDim.x * blockDim.x;
  for (size_t i = (size_t)blockIdx.x * blockDim.x + threadIdx.x; i < n; i += stride)
    out[i] = f2bf(in[i]);
}
// out[c*R + r] = bf16(in[r*C + c])   (emb (K,N) -> (N,K))
__global__ void cast_f2b_T(const float* __restrict__ in, u16* __restrict__ out, int R, int C) {
  const size_t n = (size_t)R * C;
  const size_t stride = (size_t)gridDim.x * blockDim.x;
  for (size_t i = (size_t)blockIdx.x * blockDim.x + threadIdx.x; i < n; i += stride) {
    int r = (int)(i / C), c = (int)(i % C);
    out[(size_t)c * R + r] = f2bf(in[i]);
  }
}
__global__ void copy_f32(const float* __restrict__ in, float* __restrict__ out, size_t n) {
  const size_t stride = (size_t)gridDim.x * blockDim.x;
  for (size_t i = (size_t)blockIdx.x * blockDim.x + threadIdx.x; i < n; i += stride)
    out[i] = in[i];
}

// ---------------- host orchestration ----------------
extern "C" void kernel_launch(void* const* d_in, const int* in_sizes, int n_in,
                              void* d_out, int out_size, void* d_ws, size_t ws_size,
                              hipStream_t stream) {
  (void)in_sizes; (void)n_in; (void)out_size;
  const float* x    = (const float*)d_in[0];
  const float* emb  = (const float*)d_in[1];
  const float* bn0g = (const float*)d_in[2];
  const float* bn0b = (const float*)d_in[3];
  const float* Wq   = (const float*)d_in[4];
  const float* bq   = (const float*)d_in[5];
  const float* Wk   = (const float*)d_in[6];
  const float* bk   = (const float*)d_in[7];
  const float* Wv   = (const float*)d_in[8];
  const float* bv   = (const float*)d_in[9];
  const float* bn1g = (const float*)d_in[10];
  const float* bn1b = (const float*)d_in[11];
  const float* W1   = (const float*)d_in[12];
  const float* b1   = (const float*)d_in[13];
  const float* W2   = (const float*)d_in[14];
  const float* b2   = (const float*)d_in[15];
  const float* bn2g = (const float*)d_in[16];
  const float* bn2b = (const float*)d_in[17];

  char* ws = (char*)d_ws;
  size_t off = 0;
  auto alloc = [&](size_t bytes) {
    size_t r = off;
    off += (bytes + 255) & ~(size_t)255;
    return r;
  };
  const size_t o_embT = alloc((size_t)En * En * 2);
  const size_t o_wq   = alloc((size_t)Sn * En * En * 2);
  const size_t o_wk   = alloc((size_t)Sn * En * En * 2);
  const size_t o_wv   = alloc((size_t)Sn * En * En * 2);
  const size_t o_w1   = alloc((size_t)Sn * Fn * En * 2);
  const size_t o_w2   = alloc((size_t)Sn * En * Fn * 2);
  const size_t o_xb   = alloc(BLn * En * 2);
  const size_t o_encf = alloc(BLn * En * 4);
  const size_t o_encb = alloc(BLn * En * 2);
  const size_t o_qb   = alloc(BLn * En * 2);
  const size_t o_kb   = alloc(BLn * En * 2);
  const size_t o_vb   = alloc(BLn * En * 2);
  const size_t o_sc   = alloc((size_t)HBn * LLn * 2);  // scores; later reused for FFN h1 (33.5MB<=67MB)
  const size_t o_at   = alloc((size_t)HBn * LLn * 2);  // softmaxed attn (bf16)
  const size_t o_t0   = alloc(BLn * En * 4);           // GEMM f32 out / attn out / h2
  const size_t o_of   = alloc(BLn * En * 4);           // post-attn BN out f32
  const size_t o_ob   = alloc(BLn * En * 2);
  const size_t o_st   = alloc((size_t)En * 2 * 4);
  if (ws_size < off) return;  // workspace too small: nothing safe to do

  u16*   embT = (u16*)(ws + o_embT);
  u16*   wqb  = (u16*)(ws + o_wq);
  u16*   wkb  = (u16*)(ws + o_wk);
  u16*   wvb  = (u16*)(ws + o_wv);
  u16*   w1b  = (u16*)(ws + o_w1);
  u16*   w2b  = (u16*)(ws + o_w2);
  u16*   xb   = (u16*)(ws + o_xb);
  float* encf = (float*)(ws + o_encf);
  u16*   encb = (u16*)(ws + o_encb);
  u16*   qb   = (u16*)(ws + o_qb);
  u16*   kb   = (u16*)(ws + o_kb);
  u16*   vbp  = (u16*)(ws + o_vb);
  u16*   scb  = (u16*)(ws + o_sc);
  u16*   h1b  = (u16*)(ws + o_sc);  // reuse (scores dead after softmax)
  u16*   atb  = (u16*)(ws + o_at);
  float* t0f  = (float*)(ws + o_t0);
  float* of   = (float*)(ws + o_of);
  u16*   ob   = (u16*)(ws + o_ob);
  float* st   = (float*)(ws + o_st);

  const dim3 blk256(256), blk128(128);

  // weight / input casts to bf16
  cast_f2b_T<<<dim3(1024), blk256, 0, stream>>>(emb, embT, En, En);
  cast_f2b<<<dim3(2048), blk256, 0, stream>>>(Wq, wqb, (size_t)Sn * En * En);
  cast_f2b<<<dim3(2048), blk256, 0, stream>>>(Wk, wkb, (size_t)Sn * En * En);
  cast_f2b<<<dim3(2048), blk256, 0, stream>>>(Wv, wvb, (size_t)Sn * En * En);
  cast_f2b<<<dim3(4096), blk256, 0, stream>>>(W1, w1b, (size_t)Sn * Fn * En);
  cast_f2b<<<dim3(4096), blk256, 0, stream>>>(W2, w2b, (size_t)Sn * En * Fn);
  cast_f2b<<<dim3(4096), blk256, 0, stream>>>(x, xb, BLn * En);

  const dim3 gemmE(En / 128, BLn / 128);  // N=512
  const dim3 gemmF(Fn / 128, BLn / 128);  // N=2048
  const dim3 bnStatsGrid(En / 32);
  const dim3 bnApplyGrid(4096);

  // embedding GEMM + BN0
  gemm_bf16<false><<<gemmE, blk256, 0, stream>>>(xb, embT, nullptr, t0f, nullptr,
                                                 (int)BLn, En, En);
  bn_stats<<<bnStatsGrid, blk256, 0, stream>>>(t0f, nullptr, st);
  bn_apply<<<bnApplyGrid, blk256, 0, stream>>>(t0f, nullptr, bn0g, bn0b, st, encf, encb);

  for (int i = 0; i < Sn; ++i) {
    const u16* wq_i = wqb + (size_t)i * En * En;
    const u16* wk_i = wkb + (size_t)i * En * En;
    const u16* wv_i = wvb + (size_t)i * En * En;
    const u16* w1_i = w1b + (size_t)i * Fn * En;
    const u16* w2_i = w2b + (size_t)i * En * Fn;

    // Q, K, V = relu(enc @ W^T + b)  -> bf16
    gemm_bf16<true><<<gemmE, blk256, 0, stream>>>(encb, wq_i, bq + (size_t)i * En,
                                                  nullptr, qb, (int)BLn, En, En);
    gemm_bf16<true><<<gemmE, blk256, 0, stream>>>(encb, wk_i, bk + (size_t)i * En,
                                                  nullptr, kb, (int)BLn, En, En);
    gemm_bf16<true><<<gemmE, blk256, 0, stream>>>(encb, wv_i, bv + (size_t)i * En,
                                                  nullptr, vbp, (int)BLn, En, En);

    attn_scores<<<dim3(4, 4, HBn), blk128, 0, stream>>>(qb, kb, scb);
    softmax_hb<<<dim3(256), blk256, 0, stream>>>(scb, atb);
    attn_av<<<dim3(4, HBn), blk128, 0, stream>>>(atb, vbp, t0f);

    // residual + BN1
    bn_stats<<<bnStatsGrid, blk256, 0, stream>>>(t0f, encf, st);
    bn_apply<<<bnApplyGrid, blk256, 0, stream>>>(t0f, encf, bn1g + (size_t)i * En,
                                                 bn1b + (size_t)i * En, st, of, ob);

    // FFN
    gemm_bf16<true><<<gemmF, blk256, 0, stream>>>(ob, w1_i, b1 + (size_t)i * Fn,
                                                  nullptr, h1b, (int)BLn, Fn, En);
    gemm_bf16<false><<<gemmE, blk256, 0, stream>>>(h1b, w2_i, b2 + (size_t)i * En,
                                                   t0f, nullptr, (int)BLn, En, Fn);

    // residual + BN2 -> next enc
    bn_stats<<<bnStatsGrid, blk256, 0, stream>>>(t0f, of, st);
    bn_apply<<<bnApplyGrid, blk256, 0, stream>>>(t0f, of, bn2g + (size_t)i * En,
                                                 bn2b + (size_t)i * En, st, encf, encb);
  }

  copy_f32<<<dim3(4096), blk256, 0, stream>>>(encf, (float*)d_out, BLn * En);
}
